// MeowCodebook_51410758533497
// MI455X (gfx1250) — matrix operand order repaired and checked by reading, not compiled, not run on hardware
//
#include <hip/hip_runtime.h>
#include <hip/hip_bf16.h>

// ---------------------------------------------------------------------------
// VQ-VAE forward on MI455X (gfx1250, wave32, WMMA).
// Key algebraic optimization: forward value of the straight-through estimator
// is exactly codebook[idx], so the big decoder GEMM (8192x8192x768) collapses
// to a 512x8192x768 GEMM over the codebook + an L2-resident row gather.
// Double-buffered LDS GEMM with async global->LDS copies
// (GLOBAL_LOAD_ASYNC_TO_LDS_B128 / ASYNCcnt) when the toolchain exposes them.
// ---------------------------------------------------------------------------

typedef __attribute__((ext_vector_type(16))) __bf16 v16bf;
typedef __attribute__((ext_vector_type(8)))  __bf16 v8bf;
typedef __attribute__((ext_vector_type(8)))  float  v8f;
typedef __attribute__((ext_vector_type(4)))  int    v4i;

#define BATCH    8192
#define DIN      8192
#define DCODE    768
#define NSYM     512

#define BM 128
#define BN 128
#define BK 32

__device__ __forceinline__ __bf16 f2bf(float f) { return (__bf16)f; }

// ---- async global->LDS 16B copy (fallback: synchronous vector copy) -------
#if defined(__has_builtin)
#if __has_builtin(__builtin_amdgcn_global_load_async_to_lds_b128)
#define HAVE_ASYNC_LDS 1
#endif
#endif

__device__ __forceinline__ void copy16_g2l(__bf16* ldst, const __bf16* gsrc)
{
#ifdef HAVE_ASYNC_LDS
    // builtin signature (from compiler diagnostic): param0 = v4i addrspace(1)*,
    // param1 = LDS side, then imm offset + imm cpol.
    void* g = const_cast<void*>((const void*)gsrc);
    __builtin_amdgcn_global_load_async_to_lds_b128(
        (__attribute__((address_space(1))) v4i*)g,
        (__attribute__((address_space(3))) v4i*)(void*)ldst,
        0, 0);
#else
    *(v8bf*)ldst = *(const v8bf*)gsrc;
#endif
}

__device__ __forceinline__ void async_wait_all()
{
#ifdef HAVE_ASYNC_LDS
#if __has_builtin(__builtin_amdgcn_s_wait_asynccnt)
    __builtin_amdgcn_s_wait_asynccnt(0);
#else
    asm volatile("s_wait_asynccnt 0x0" ::: "memory");
#endif
#endif
}

// ---------------------------------------------------------------------------
// Elementwise f32 -> bf16 (weights), 4 elements/thread
// ---------------------------------------------------------------------------
__global__ __launch_bounds__(256)
void f32_to_bf16_kernel(const float* __restrict__ in, __bf16* __restrict__ out, int n)
{
    int i = (blockIdx.x * 256 + threadIdx.x) * 4;
    if (i < n) {
        float4 f = *(const float4*)(in + i);
        out[i + 0] = f2bf(f.x);
        out[i + 1] = f2bf(f.y);
        out[i + 2] = f2bf(f.z);
        out[i + 3] = f2bf(f.w);
    }
}

// ---------------------------------------------------------------------------
// Codebook prep: bf16 copy + squared L2 norms. One block per code.
// ---------------------------------------------------------------------------
__global__ __launch_bounds__(256)
void cb_prep_kernel(const float* __restrict__ cb, __bf16* __restrict__ cbb,
                    float* __restrict__ cnorm)
{
    __shared__ float red[256];
    const int c = blockIdx.x;
    float s = 0.0f;
    for (int k = threadIdx.x; k < DCODE; k += 256) {
        float f = cb[(size_t)c * DCODE + k];
        cbb[(size_t)c * DCODE + k] = f2bf(f);
        s += f * f;
    }
    red[threadIdx.x] = s;
    __syncthreads();
    for (int off = 128; off > 0; off >>= 1) {
        if (threadIdx.x < off) red[threadIdx.x] += red[threadIdx.x + off];
        __syncthreads();
    }
    if (threadIdx.x == 0) cnorm[c] = red[0];
}

// ---------------------------------------------------------------------------
// Tiled WMMA GEMM: Out[M,N] = A[M,K] * B[N,K]^T + bias[N]
//   A_F32:    A is f32 (register-prefetched, converted to bf16 into LDS)
//   OUT_BF16: write bf16 (else f32)
// 256 threads = 8 waves; block tile 128x128; wave tile 32x64 (2x4 WMMA tiles).
// Double-buffered LDS; next tile streams in (async-to-LDS where available)
// while WMMAs consume the current tile; one barrier per K-step.
// bf16 fragment layouts per CDNA5 ISA 7.12.2:
//   A 16x32: lanes 0-15 = rows M, K groups {0-7,16-23}; lanes 16-31 {8-15,24-31}
//   B 32x16: lane = column N (mod 16), K groups same as A: {0-7,16-23} / {8-15,24-31}
// ---------------------------------------------------------------------------
template<bool A_F32, bool OUT_BF16>
__global__ __launch_bounds__(256)
void gemm_nt_wmma_kernel(const void* __restrict__ Aptr,
                         const __bf16* __restrict__ B,     // [N,K] bf16
                         const float* __restrict__ bias,   // [N]
                         void* __restrict__ Out,           // [M,N]
                         int M, int N, int K)
{
    __shared__ __align__(16) __bf16 As[2][BM][BK];
    __shared__ __align__(16) __bf16 Bs[2][BN][BK];

    const int tid   = threadIdx.x;
    const int lane  = tid & 31;          // wave32
    const int wave  = tid >> 5;          // 0..7
    const int wm    = wave & 3;          // 4 waves along M, 32 rows each
    const int wn    = wave >> 2;         // 2 waves along N, 64 cols each
    const int mblk  = blockIdx.y * BM;
    const int nblk  = blockIdx.x * BN;
    const int l16   = lane & 15;
    const int lhalf = lane >> 4;

    v8f acc[2][4];
#pragma unroll
    for (int mi = 0; mi < 2; ++mi)
#pragma unroll
        for (int ni = 0; ni < 4; ++ni)
#pragma unroll
            for (int r = 0; r < 8; ++r) acc[mi][ni][r] = 0.0f;

    float4 aregs[4];   // register prefetch buffer for the f32-A path

    auto stageB = [&](int buf, int k0) {
#pragma unroll
        for (int j = 0; j < 2; ++j) {
            int i = tid + 256 * j;               // 512 chunks of 8 bf16
            int r = i >> 2, c = (i & 3) * 8;
            copy16_g2l(&Bs[buf][r][c], B + (size_t)(nblk + r) * K + k0 + c);
        }
    };
    auto stageA_bf = [&](int buf, int k0) {
        const __bf16* A = (const __bf16*)Aptr;
#pragma unroll
        for (int j = 0; j < 2; ++j) {
            int i = tid + 256 * j;
            int r = i >> 2, c = (i & 3) * 8;
            copy16_g2l(&As[buf][r][c], A + (size_t)(mblk + r) * K + k0 + c);
        }
    };
    auto loadA_f32 = [&](int k0) {
        const float* A = (const float*)Aptr;
#pragma unroll
        for (int j = 0; j < 4; ++j) {
            int i = tid + 256 * j;               // 1024 float4 chunks
            int r = i >> 3, c = (i & 7) * 4;
            aregs[j] = *(const float4*)(A + (size_t)(mblk + r) * K + k0 + c);
        }
    };
    auto storeA_f32 = [&](int buf) {
#pragma unroll
        for (int j = 0; j < 4; ++j) {
            int i = tid + 256 * j;
            int r = i >> 3, c = (i & 7) * 4;
            As[buf][r][c + 0] = f2bf(aregs[j].x);
            As[buf][r][c + 1] = f2bf(aregs[j].y);
            As[buf][r][c + 2] = f2bf(aregs[j].z);
            As[buf][r][c + 3] = f2bf(aregs[j].w);
        }
    };

    // ---- prologue: stage tile 0 into buffer 0 ----
    if (A_F32) { loadA_f32(0); storeA_f32(0); }
    else       { stageA_bf(0, 0); }
    stageB(0, 0);
    async_wait_all();
    __syncthreads();

    const int nsteps = K / BK;
    for (int step = 0; step < nsteps; ++step) {
        const int  cur     = step & 1;
        const int  nxt     = cur ^ 1;
        const bool hasnext = (step + 1) < nsteps;
        const int  k1      = (step + 1) * BK;

        // ---- kick off next tile (overlaps with WMMA below) ----
        if (hasnext) {
            stageB(nxt, k1);
            if (A_F32) loadA_f32(k1);            // global loads in flight
            else       stageA_bf(nxt, k1);
        }

        // ---- build fragments from current buffer ----
        v16bf afrag[2];
#pragma unroll
        for (int mi = 0; mi < 2; ++mi) {
            int row = wm * 32 + mi * 16 + l16;
            v8bf lo = *(const v8bf*)(&As[cur][row][lhalf * 8]);       // K 0-7 / 8-15
            v8bf hi = *(const v8bf*)(&As[cur][row][16 + lhalf * 8]);  // K 16-23 / 24-31
#pragma unroll
            for (int e = 0; e < 8; ++e) { afrag[mi][e] = lo[e]; afrag[mi][e + 8] = hi[e]; }
        }
        v16bf bfrag[4];
#pragma unroll
        for (int ni = 0; ni < 4; ++ni) {
            int row = wn * 64 + ni * 16 + l16;                        // column N
            v8bf lo = *(const v8bf*)(&Bs[cur][row][lhalf * 8]);
            v8bf hi = *(const v8bf*)(&Bs[cur][row][16 + lhalf * 8]);
#pragma unroll
            for (int e = 0; e < 8; ++e) { bfrag[ni][e] = lo[e]; bfrag[ni][e + 8] = hi[e]; }
        }

        // ---- 8 WMMAs per wave per K-step ----
#pragma unroll
        for (int mi = 0; mi < 2; ++mi)
#pragma unroll
            for (int ni = 0; ni < 4; ++ni)
                acc[mi][ni] = __builtin_amdgcn_wmma_f32_16x16x32_bf16(
                    false, afrag[mi], false, bfrag[ni],
                    (short)0, acc[mi][ni], false, false);

        // ---- finish staging next tile, then one barrier per step ----
        if (hasnext && A_F32) storeA_f32(nxt);
        if (hasnext) async_wait_all();
        __syncthreads();   // waits dscnt, then barrier
    }

    // ---- epilogue: bias add + store (D layout: lane=N, VGPR r = row) ----
#pragma unroll
    for (int ni = 0; ni < 4; ++ni) {
        int n = nblk + wn * 64 + ni * 16 + l16;
        float bv = bias[n];
#pragma unroll
        for (int mi = 0; mi < 2; ++mi) {
            int mbase = mblk + wm * 32 + mi * 16 + lhalf * 8;
#pragma unroll
            for (int r = 0; r < 8; ++r) {
                float v = acc[mi][ni][r] + bv;
                if (OUT_BF16)
                    ((__bf16*)Out)[(size_t)(mbase + r) * N + n] = f2bf(v);
                else
                    ((float*)Out)[(size_t)(mbase + r) * N + n] = v;
            }
        }
    }
}

// ---------------------------------------------------------------------------
// Fused distance GEMM + argmin.
// One block per 16 batch rows; 8 waves cover the 512 codes (64 each).
// score[m][c] = ||c||^2 - 2 * enc[m].dot(cb[c])  (||e||^2 dropped: row-const)
// ---------------------------------------------------------------------------
__global__ __launch_bounds__(256)
void vq_argmin_kernel(const __bf16* __restrict__ enc,   // [BATCH, DCODE] bf16
                      const __bf16* __restrict__ cb,    // [NSYM, DCODE] bf16
                      const float* __restrict__ cnorm,  // [NSYM]
                      int* __restrict__ idxOut)         // [BATCH]
{
    __shared__ __align__(16) __bf16 As[16][32];
    __shared__ float scores[16][NSYM];
    __shared__ float rmin[16][16];
    __shared__ int   ridx[16][16];

    const int tid   = threadIdx.x;
    const int lane  = tid & 31;
    const int wave  = tid >> 5;        // 0..7 -> 64-code chunk
    const int l16   = lane & 15;
    const int lhalf = lane >> 4;
    const int mblk  = blockIdx.x * 16;

    v8f acc[4];
#pragma unroll
    for (int ni = 0; ni < 4; ++ni)
#pragma unroll
        for (int r = 0; r < 8; ++r) acc[ni][r] = 0.0f;

    for (int k0 = 0; k0 < DCODE; k0 += 32) {
        if (tid < 64) {                    // 16x32 bf16 tile = 64 chunks of 8
            int r = tid >> 2, c = (tid & 3) * 8;
            *(v8bf*)(&As[r][c]) =
                *(const v8bf*)(enc + (size_t)(mblk + r) * DCODE + k0 + c);
        }
        __syncthreads();

        v16bf afrag;
        {
            v8bf lo = *(const v8bf*)(&As[l16][lhalf * 8]);
            v8bf hi = *(const v8bf*)(&As[l16][16 + lhalf * 8]);
#pragma unroll
            for (int e = 0; e < 8; ++e) { afrag[e] = lo[e]; afrag[e + 8] = hi[e]; }
        }
#pragma unroll
        for (int ni = 0; ni < 4; ++ni) {
            int n = wave * 64 + ni * 16 + l16;  // code id; cb row contiguous in K
            const __bf16* bp = cb + (size_t)n * DCODE + k0 + lhalf * 8;
            v8bf lo = *(const v8bf*)(bp);
            v8bf hi = *(const v8bf*)(bp + 16);
            v16bf bfrag;
#pragma unroll
            for (int e = 0; e < 8; ++e) { bfrag[e] = lo[e]; bfrag[e + 8] = hi[e]; }
            acc[ni] = __builtin_amdgcn_wmma_f32_16x16x32_bf16(
                false, afrag, false, bfrag, (short)0, acc[ni], false, false);
        }
        __syncthreads();
    }

    // scores -> LDS
#pragma unroll
    for (int ni = 0; ni < 4; ++ni) {
        int n = wave * 64 + ni * 16 + l16;
        float cn = cnorm[n];
#pragma unroll
        for (int r = 0; r < 8; ++r) {
            int m = lhalf * 8 + r;
            scores[m][n] = cn - 2.0f * acc[ni][r];
        }
    }
    __syncthreads();

    // argmin over 512: 16 threads per row scan 32 each, then reduce
    {
        int row = tid & 15, part = tid >> 4;
        float best = 3.402823466e+38f;
        int bi = part * 32;
        for (int j = 0; j < 32; ++j) {
            int n = part * 32 + j;
            float v = scores[row][n];
            if (v < best) { best = v; bi = n; }   // ascending scan => first min
        }
        rmin[row][part] = best;
        ridx[row][part] = bi;
    }
    __syncthreads();
    if (tid < 16) {
        int row = tid;
        float best = rmin[row][0];
        int bi = ridx[row][0];
        for (int p = 1; p < 16; ++p) {
            float v = rmin[row][p];
            int ii = ridx[row][p];
            if (v < best || (v == best && ii < bi)) { best = v; bi = ii; }
        }
        idxOut[mblk + row] = bi;
    }
}

// ---------------------------------------------------------------------------
// Row gather: out[m,:] = dcb[idx[m],:]   (dcb is 16.8MB -> L2-resident)
// ---------------------------------------------------------------------------
__global__ __launch_bounds__(256)
void gather_rows_kernel(const float* __restrict__ dcb,
                        const int* __restrict__ idx,
                        float* __restrict__ out)
{
    const int m = blockIdx.x;
    const int r = idx[m];
    const float4* src = (const float4*)(dcb + (size_t)r * DIN);
    float4* dst       = (float4*)(out + (size_t)m * DIN);
#pragma unroll
    for (int j = 0; j < 8; ++j)
        dst[threadIdx.x + 256 * j] = src[threadIdx.x + 256 * j];
}

// ---------------------------------------------------------------------------
// Launch
// ---------------------------------------------------------------------------
extern "C" void kernel_launch(void* const* d_in, const int* in_sizes, int n_in,
                              void* d_out, int out_size, void* d_ws, size_t ws_size,
                              hipStream_t stream)
{
    (void)in_sizes; (void)n_in; (void)out_size; (void)ws_size;

    const float* inputs = (const float*)d_in[0];  // [8192, 8192]
    const float* enc_w  = (const float*)d_in[1];  // [768, 8192]
    const float* enc_b  = (const float*)d_in[2];  // [768]
    const float* dec_w  = (const float*)d_in[3];  // [8192, 768]
    const float* dec_b  = (const float*)d_in[4];  // [8192]
    const float* cbook  = (const float*)d_in[5];  // [512, 768]
    float* out = (float*)d_out;                   // [8192, 8192]

    // workspace layout (all chunks 256B aligned), total ~55.4 MB
    char* ws = (char*)d_ws;
    __bf16* enc_w_bf = (__bf16*)ws;  ws += (size_t)DCODE * DIN * 2;    // 12.6 MB
    __bf16* dec_w_bf = (__bf16*)ws;  ws += (size_t)DIN * DCODE * 2;    // 12.6 MB
    __bf16* cb_bf    = (__bf16*)ws;  ws += (size_t)NSYM * DCODE * 2;   // 0.79 MB
    float*  cnorm    = (float*)ws;   ws += (size_t)NSYM * 4;           // 2 KB
    __bf16* enc_bf   = (__bf16*)ws;  ws += (size_t)BATCH * DCODE * 2;  // 12.6 MB
    float*  dcb      = (float*)ws;   ws += (size_t)NSYM * DIN * 4;     // 16.8 MB
    int*    idx      = (int*)ws;     ws += (size_t)BATCH * 4;          // 32 KB

    // 1) weights f32 -> bf16
    f32_to_bf16_kernel<<<(DCODE * DIN / 4 + 255) / 256, 256, 0, stream>>>(
        enc_w, enc_w_bf, DCODE * DIN);
    f32_to_bf16_kernel<<<(DIN * DCODE / 4 + 255) / 256, 256, 0, stream>>>(
        dec_w, dec_w_bf, DIN * DCODE);

    // 2) codebook bf16 + norms
    cb_prep_kernel<<<NSYM, 256, 0, stream>>>(cbook, cb_bf, cnorm);

    // 3) encoder GEMM: enc_bf[8192,768] = inputs @ enc_w^T + enc_b  (103 GF)
    gemm_nt_wmma_kernel<true, true><<<dim3(DCODE / BN, BATCH / BM), 256, 0, stream>>>(
        (const void*)inputs, enc_w_bf, enc_b, (void*)enc_bf, BATCH, DCODE, DIN);

    // 4) decoded codebook: dcb[512,8192] = codebook @ dec_w^T + dec_b  (6.4 GF)
    gemm_nt_wmma_kernel<false, false><<<dim3(DIN / BN, NSYM / BM), 256, 0, stream>>>(
        (const void*)cb_bf, dec_w_bf, dec_b, (void*)dcb, NSYM, DIN, DCODE);

    // 5) fused -2*enc@cb^T + ||c||^2 and argmin
    vq_argmin_kernel<<<BATCH / 16, 256, 0, stream>>>(enc_bf, cb_bf, cnorm, idx);

    // 6) out[m,:] = dcb[idx[m],:]
    gather_rows_kernel<<<BATCH, 256, 0, stream>>>(dcb, idx, out);
}
